// FourierFeatureMLP_824633721577
// MI455X (gfx1250) — matrix operand order repaired and checked by reading, not compiled, not run on hardware
//
#include <hip/hip_runtime.h>

typedef _Float16 v16h __attribute__((ext_vector_type(16)));
typedef _Float16 v8h  __attribute__((ext_vector_type(8)));
typedef float    v8f  __attribute__((ext_vector_type(8)));

#define HD   256   // hidden dim
#define NF   128   // num freqs (2*NF == HD)
#define NLAY 8     // input layer + 7 hidden layers (all HD x HD GEMMs)
#define ROWS 256   // rows per workgroup (8 waves x 32 rows each)
#define CHUNK_HALFS (128 * HD)          // 128 W-rows x 256 K = 64 KB f16
#define NPHASE (NLAY * 2)               // 2 weight chunks per layer

// LDS layout (bytes from dynamic-shared base)
#define OFF_W0   0          // 64 KB  weight ping buffer
#define OFF_W1   65536      // 64 KB  weight pong buffer
#define OFF_ACT  131072     // 128 KB activations (256 x 256 f16)
#define OFF_BIAS 262144     // 8 KB   biases [8][256] f32
#define OFF_WOUT 270336     // 1 KB+4 W_out f32 + b_out
#define LDS_TOTAL (OFF_WOUT + (HD + 1) * 4)

// ---------------------------------------------------------------------------
// Hardware tanh (gfx1250 V_TANH_F32) -- builtin confirmed to lower correctly.
// ---------------------------------------------------------------------------
__device__ __forceinline__ float fast_tanh(float x) {
#if __has_builtin(__builtin_amdgcn_tanhf)
    return __builtin_amdgcn_tanhf(x);
#else
    float r;
    asm volatile("v_tanh_f32 %0, %1\n\tv_nop\n\tv_nop" : "=v"(r) : "v"(x));
    return r;
#endif
}

// ---------------------------------------------------------------------------
// Async global->LDS 16B copy (per lane), tracked by ASYNCcnt.
// ---------------------------------------------------------------------------
__device__ __forceinline__ void async_copy_b128(unsigned lds_off,
                                                unsigned long long gaddr) {
    asm volatile("global_load_async_to_lds_b128 %0, %1, off"
                 :: "v"(lds_off), "v"(gaddr) : "memory");
}
__device__ __forceinline__ void wait_asynccnt0() {
    asm volatile("s_wait_asynccnt 0x0" ::: "memory");
}

// ---------------------------------------------------------------------------
// Kernel 1: convert all matmul weights to f16, packed [8][256][256] in d_ws.
// ---------------------------------------------------------------------------
__global__ void convert_weights_f16(const float* __restrict__ W_in,
                                    const float* __restrict__ W_h,
                                    _Float16* __restrict__ out) {
    int idx = blockIdx.x * blockDim.x + threadIdx.x;
    const int per = HD * HD;
    if (idx < NLAY * per) {
        float v = (idx < per) ? W_in[idx] : W_h[idx - per];
        out[idx] = (_Float16)v;
    }
}

union AFrag { v16h v; v8h h[2]; };

__device__ __forceinline__ v8f wmma_f16(const AFrag& a, const AFrag& b, v8f c) {
    return __builtin_amdgcn_wmma_f32_16x16x32_f16(
             false, a.v, false, b.v, (short)0, c, false, false);
}

// ---------------------------------------------------------------------------
// One weight-chunk compute: 8 col-tiles x 8 K-steps, TWO M-tiles per wave so
// each B fragment feeds two independent WMMA chains. B is double-buffered in
// registers (load ks+1 while ks computes). wchunk must be a constant-offset
// LDS pointer at every call site so clang emits ds_load_b128.
// ---------------------------------------------------------------------------
__device__ __forceinline__ void compute_chunk(
    const _Float16* wchunk,       // 128 W-rows x 256 K, in LDS
    _Float16* actLDS,             // 256 x 256 activations, in LDS
    const float* bAll,            // [8][256] biases, in LDS
    const AFrag (&a0)[8], const AFrag (&a1)[8],
    int layer, int c, int wv, int l16, int halfId)
{
    for (int t = 0; t < 8; ++t) {
        const int colBase = c * 128 + t * 16;
        const float bv = bAll[layer * HD + colBase + l16];
        v8f acc0, acc1;
        #pragma unroll
        for (int i = 0; i < 8; ++i) { acc0[i] = bv; acc1[i] = bv; }

        // B[k][n] = W[n][k]; chunk holds 128 full W rows, stride 256 halfs.
        const _Float16* wrow = wchunk + (t * 16 + l16) * HD;
        AFrag b[2];
        b[0].h[0] = *(const v8h*)(wrow + 8 * halfId);
        b[0].h[1] = *(const v8h*)(wrow + 16 + 8 * halfId);
        #pragma unroll
        for (int ks = 0; ks < 8; ++ks) {
            const int cur = ks & 1, nxt = cur ^ 1;
            if (ks < 7) {   // prefetch next K-step's B while computing
                b[nxt].h[0] = *(const v8h*)(wrow + (ks + 1) * 32 + 8 * halfId);
                b[nxt].h[1] = *(const v8h*)(wrow + (ks + 1) * 32 + 16 + 8 * halfId);
            }
            acc0 = wmma_f16(a0[ks], b[cur], acc0);
            acc1 = wmma_f16(a1[ks], b[cur], acc1);
        }

        // Batch all tanh (TRANS ops pipeline back-to-back), then cvt+store.
        float th0[8], th1[8];
        #pragma unroll
        for (int r = 0; r < 8; ++r) { th0[r] = fast_tanh(acc0[r]); }
        #pragma unroll
        for (int r = 0; r < 8; ++r) { th1[r] = fast_tanh(acc1[r]); }

        // C/D layout: VGPR r -> row M = r + 8*halfId, col N = l16.
        _Float16* d0 = actLDS + (wv * 32 + 8 * halfId) * HD + colBase + l16;
        _Float16* d1 = d0 + 16 * HD;
        #pragma unroll
        for (int r = 0; r < 8; ++r) { d0[r * HD] = (_Float16)th0[r]; }
        #pragma unroll
        for (int r = 0; r < 8; ++r) { d1[r * HD] = (_Float16)th1[r]; }
    }
}

// ---------------------------------------------------------------------------
// Kernel 2: fully fused Fourier-feature MLP with async-DMA weight pipeline.
// Block = 256 threads (8 wave32), 256 rows per block, activations in LDS.
// ---------------------------------------------------------------------------
__global__ __launch_bounds__(256)
void fused_ffmlp(const float* __restrict__ tx,
                 const float* __restrict__ Bf,      // [F][2]
                 const _Float16* __restrict__ Wall, // [8][256][256] f16
                 const float* __restrict__ b_in,    // [256]
                 const float* __restrict__ b_h,     // [7][256]
                 const float* __restrict__ W_out,   // [1][256]
                 const float* __restrict__ b_out,   // [1]
                 float* __restrict__ out) {
    extern __shared__ char smem[];
    _Float16* wbuf0  = (_Float16*)(smem + OFF_W0);
    _Float16* wbuf1  = (_Float16*)(smem + OFF_W1);
    _Float16* actLDS = (_Float16*)(smem + OFF_ACT);
    float*    bAll   = (float*)   (smem + OFF_BIAS);
    float*    woutL  = (float*)   (smem + OFF_WOUT);

    const int tid    = threadIdx.x;
    const int lane   = tid & 31;
    const int wv     = tid >> 5;       // wave id 0..7 -> rows 32*wv..32*wv+31
    const int halfId = lane >> 4;
    const int l16    = lane & 15;
    const int r0     = blockIdx.x * ROWS;

    const unsigned ldsW0 = (unsigned)(size_t)wbuf0;
    const unsigned ldsW1 = (unsigned)(size_t)wbuf1;

    // Prologue A: kick off async DMA of weight chunk 0 into wbuf0.
    {
        const unsigned long long g0 = (unsigned long long)(const char*)Wall;
        #pragma unroll
        for (int i = 0; i < 16; ++i) {
            unsigned ofs = (unsigned)(tid * 16 + i * 4096);  // 16B per lane
            async_copy_b128(ldsW0 + ofs, g0 + ofs);
        }
    }

    // Prologue B: stage ALL biases once ([8][256] f32) + W_out/b_out.
    for (int i = tid; i < NLAY * HD; i += 256) {
        int layer = i >> 8, col = i & (HD - 1);
        bAll[i] = (layer == 0) ? b_in[col] : b_h[(layer - 1) * HD + col];
    }
    woutL[tid] = W_out[tid];
    if (tid == 0) woutL[HD] = b_out[0];

    // Prologue C: Fourier features -> actLDS (f16, row-major, stride 256).
    for (int p = tid; p < ROWS * NF; p += 256) {
        int r = p >> 7;
        int f = p & (NF - 1);
        float x0 = tx[(r0 + r) * 2 + 0];
        float x1 = tx[(r0 + r) * 2 + 1];
        float pr = 6.283185307179586f * (x0 * Bf[f * 2 + 0] + x1 * Bf[f * 2 + 1]);
        actLDS[r * HD + f]      = (_Float16)__sinf(pr);
        actLDS[r * HD + NF + f] = (_Float16)__cosf(pr);
    }

    wait_asynccnt0();
    __syncthreads();

    // Main pipeline: 16 phases (8 layers x 2 weight chunks of 128 out-cols).
    // Phase protocol: issue DMA into idle buffer -> compute from live buffer
    //                 -> wait own async -> block barrier.
    AFrag a0[8], a1[8];
    for (int ph = 0; ph < NPHASE; ++ph) {
        const int layer = ph >> 1;
        const int c     = ph & 1;                 // chunk (== buffer parity)

        // Issue DMA of the next chunk into the buffer everyone just vacated.
        if (ph + 1 < NPHASE) {
            const unsigned long long g =
                (unsigned long long)(const char*)(Wall + (size_t)(ph + 1) * CHUNK_HALFS);
            const unsigned ldsn = (c == 0) ? ldsW1 : ldsW0;
            #pragma unroll
            for (int i = 0; i < 16; ++i) {
                unsigned ofs = (unsigned)(tid * 16 + i * 4096);
                async_copy_b128(ldsn + ofs, g + ofs);
            }
        }

        // Load A fragments once per layer: both 16-row M-tiles of this wave.
        // A 16x32 layout: lanes 0-15 row M=lane hold K [0..7]&[16..23],
        // lanes 16-31 row M=lane-16 hold K [8..15]&[24..31].
        if (c == 0) {
            const _Float16* ar0 = actLDS + (wv * 32 + l16) * HD;
            const _Float16* ar1 = ar0 + 16 * HD;
            #pragma unroll
            for (int ks = 0; ks < 8; ++ks) {
                a0[ks].h[0] = *(const v8h*)(ar0 + ks * 32 + 8 * halfId);
                a0[ks].h[1] = *(const v8h*)(ar0 + ks * 32 + 16 + 8 * halfId);
                a1[ks].h[0] = *(const v8h*)(ar1 + ks * 32 + 8 * halfId);
                a1[ks].h[1] = *(const v8h*)(ar1 + ks * 32 + 16 + 8 * halfId);
            }
        }

        // Constant-offset LDS pointers at each call site -> ds_load_b128.
        if (c == 0) {
            compute_chunk(wbuf0, actLDS, bAll, a0, a1, layer, 0, wv, l16, halfId);
        } else {
            compute_chunk(wbuf1, actLDS, bAll, a0, a1, layer, 1, wv, l16, halfId);
        }

        wait_asynccnt0();
        __syncthreads();
    }

    // Output layer: y = h . W_out + b_out, one row per thread, all from LDS.
    {
        const _Float16* hrow = actLDS + tid * HD;
        float s = woutL[HD];
        #pragma unroll 8
        for (int k = 0; k < HD; ++k) s += (float)hrow[k] * woutL[k];
        out[r0 + tid] = s;
    }
}

// ---------------------------------------------------------------------------
extern "C" void kernel_launch(void* const* d_in, const int* in_sizes, int n_in,
                              void* d_out, int out_size, void* d_ws, size_t ws_size,
                              hipStream_t stream) {
    const float* tx    = (const float*)d_in[0];
    const float* Bf    = (const float*)d_in[1];
    const float* W_in  = (const float*)d_in[2];
    const float* b_in  = (const float*)d_in[3];
    const float* W_h   = (const float*)d_in[4];
    const float* b_h   = (const float*)d_in[5];
    const float* W_out = (const float*)d_in[6];
    const float* b_out = (const float*)d_in[7];

    _Float16* Wf16 = (_Float16*)d_ws;   // 8*256*256 halfs = 1 MB scratch

    const int totalW = NLAY * HD * HD;
    convert_weights_f16<<<(totalW + 255) / 256, 256, 0, stream>>>(W_in, W_h, Wf16);

    const int n = in_sizes[0] / 2;      // N collocation points
    const int blocks = n / ROWS;        // 131072/256 = 512
    fused_ffmlp<<<blocks, 256, LDS_TOTAL, stream>>>(tx, Bf, Wf16, b_in, b_h,
                                                    W_out, b_out, (float*)d_out);
}
